// E1Cell_8151847928498
// MI455X (gfx1250) — compile-verified, matrix operand order checked by reading, not run on hardware
//
#include <hip/hip_runtime.h>
#include <hip/hip_bf16.h>
#include <math.h>

// CDNA5 (gfx1250) persistent-kernel Elman recurrence with WMMA.
// h_t = tanh(W_x x_t + W_h h_{t-1} + b); out_t = h_t * silu(x_t)
//
// 64 persistent WGs x 128 threads (4 wave32). Each WG owns 16 output columns;
// its W_x / W_h column slices live in LDS as bf16 for the whole kernel.
// Per step, each wave computes a K=256 quarter of both GEMM fragments with
// v_wmma_f32_16x16x32_bf16 (M padded 8->16, f32 accumulate). The x-projection
// for step t+1 is computed off the recurrent critical path. Cross-WG h
// broadcast via ping-pong bf16 buffer + per-step counter-slot grid barrier.

typedef __attribute__((ext_vector_type(16))) __bf16 v16bf;
typedef __attribute__((ext_vector_type(8)))  float  v8f;

#define B_  8
#define T_  4096
#define D_  1024
#define G_  64          // persistent workgroups, each owns NS output columns
#define NS  16
#define NW  4           // waves per workgroup (wave32)
#define NT  (NW*32)

#define WS_K 1032       // padded bf16 stride for LDS weight rows (bank-conflict-free)

__device__ __forceinline__ int kpair(int p, int lh) {
    // K offset of dword-pair p for 16-bit WMMA operand layout (lane half lh)
    return ((p < 4) ? (2*p) : (2*(p-4) + 16)) + 8*lh;
}

union BU { unsigned u[8]; v16bf v; };

// B-matrix fragment (K x 16) from LDS-resident bf16 weight slice
__device__ __forceinline__ v16bf load_wfrag(const __bf16* w, int n, int kbase, int lh) {
    const unsigned* w32 = (const unsigned*)w;
    BU f;
#pragma unroll
    for (int p = 0; p < 8; ++p) {
        int k = kbase + kpair(p, lh);
        f.u[p] = w32[(n * WS_K + k) >> 1];
    }
    return f.v;
}

// A-matrix fragment from global bf16 h buffer (row stride D_, rows >= B_ zero)
__device__ __forceinline__ v16bf load_hfrag_g(const unsigned* hb32, int m, int kbase, int lh) {
    BU f;
    if (m < B_) {
#pragma unroll
        for (int p = 0; p < 8; ++p) {
            int k = kbase + kpair(p, lh);
            f.u[p] = hb32[m * (D_ / 2) + (k >> 1)];
        }
    } else {
#pragma unroll
        for (int p = 0; p < 8; ++p) f.u[p] = 0u;
    }
    return f.v;
}

// A-matrix fragment from global f32 with given row stride, converted to bf16
__device__ __forceinline__ v16bf load_f32frag_g(const float* base, size_t rstride,
                                                int m, int kbase, int lh) {
    v16bf a;
    if (m < B_) {
        const float* r = base + (size_t)m * rstride + kbase;
#pragma unroll
        for (int p = 0; p < 8; ++p) {
            int k = kpair(p, lh);
            a[2*p]   = (__bf16)r[k];
            a[2*p+1] = (__bf16)r[k + 1];
        }
    } else {
#pragma unroll
        for (int i = 0; i < 16; ++i) a[i] = (__bf16)0.0f;
    }
    return a;
}

__global__ void init_cnt_kernel(unsigned* cnt) {
    int i = blockIdx.x * blockDim.x + threadIdx.x;
    if (i < T_) cnt[i] = 0u;
}

__global__ void __launch_bounds__(NT, 1)
elman_persistent_kernel(const float* __restrict__ x,  const float* __restrict__ h0,
                        const float* __restrict__ Wx, const float* __restrict__ Wh,
                        const float* __restrict__ b,  float* __restrict__ out,
                        unsigned* __restrict__ cnt,   __bf16* __restrict__ hbuf)
{
    __shared__ __bf16 sWx[NS * WS_K];     // W_x column slice, bf16 (~33 KB)
    __shared__ __bf16 sWh[NS * WS_K];     // W_h column slice, bf16 (~33 KB)
    __shared__ float  sC[NW][256];        // per-wave C partials
    __shared__ float  sB[NS];

    const int wg   = blockIdx.x;
    const int tid  = threadIdx.x;
    const int w    = tid >> 5;
    const int lane = tid & 31;
    const int lh   = lane >> 4;
    const int ln   = lane & 15;           // N for B/C layout, M for A layout
    const int col0 = wg * NS;
    const v8f vzero = {};

    // ---- prologue: weights + bias resident in LDS for the whole kernel ----
    for (int i = tid; i < NS * D_; i += NT) {
        int n = i >> 10, k = i & (D_ - 1);
        sWx[n * WS_K + k] = (__bf16)Wx[(size_t)(col0 + n) * D_ + k];
        sWh[n * WS_K + k] = (__bf16)Wh[(size_t)(col0 + n) * D_ + k];
    }
    if (tid < NS) sB[tid] = b[col0 + tid];
    __syncthreads();

    const int k0 = w * (D_ / NW);         // this wave's K range (256 wide)

    // x-projection partial for t=0 (off the recurrent critical path)
    v8f acc = vzero;
#pragma unroll
    for (int kt = 0; kt < D_ / NW; kt += 32) {
        v16bf a  = load_f32frag_g(x, (size_t)T_ * D_, ln, k0 + kt, lh);
        v16bf bb = load_wfrag(sWx, ln, k0 + kt, lh);
        acc = __builtin_amdgcn_wmma_f32_16x16x32_bf16(false, a, false, bb,
                                                      (short)0, acc, false, false);
    }

    for (int t = 0; t < T_; ++t) {
        // ---- recurrent GEMM: h_{t-1} @ W_h^T accumulated onto xw partial ----
        if (t == 0) {
            // h0 comes straight from the f32 input (once)
#pragma unroll
            for (int kt = 0; kt < D_ / NW; kt += 32) {
                v16bf a  = load_f32frag_g(h0, (size_t)D_, ln, k0 + kt, lh);
                v16bf bb = load_wfrag(sWh, ln, k0 + kt, lh);
                acc = __builtin_amdgcn_wmma_f32_16x16x32_bf16(false, a, false, bb,
                                                              (short)0, acc, false, false);
            }
        } else {
            // one acquire-spin per wave: each CU invalidates its own L0
            if (lane == 0) {
                while (__hip_atomic_load(&cnt[t-1], __ATOMIC_ACQUIRE,
                                         __HIP_MEMORY_SCOPE_AGENT) < G_)
                    __builtin_amdgcn_s_sleep(1);
            }
            const unsigned* hb32 =
                (const unsigned*)(hbuf + (size_t)((t-1) & 1) * (B_ * D_));
#pragma unroll
            for (int kt = 0; kt < D_ / NW; kt += 32) {
                v16bf a  = load_hfrag_g(hb32, ln, k0 + kt, lh);
                v16bf bb = load_wfrag(sWh, ln, k0 + kt, lh);
                acc = __builtin_amdgcn_wmma_f32_16x16x32_bf16(false, a, false, bb,
                                                              (short)0, acc, false, false);
            }
        }

#pragma unroll
        for (int v = 0; v < 8; ++v) sC[w][v * 32 + lane] = acc[v];
        __syncthreads();

        // ---- wave 0: reduce, tanh, publish h_t, gate output ----
        if (w == 0) {
            float hv[8];
#pragma unroll
            for (int v = 0; v < 8; ++v) {
                float s = sC[0][v*32 + lane] + sC[1][v*32 + lane]
                        + sC[2][v*32 + lane] + sC[3][v*32 + lane] + sB[ln];
                hv[v] = tanhf(s);
            }
            __bf16* hw = hbuf + (size_t)(t & 1) * (B_ * D_);
            if (lane < 16) {
#pragma unroll
                for (int v = 0; v < 8; ++v)                // rows m = v (B_=8 valid)
                    hw[v * D_ + col0 + lane] = (__bf16)hv[v];
            }
            __threadfence();
            if (lane == 0)   // release other WGPs before doing our output stores
                __hip_atomic_fetch_add(&cnt[t], 1u, __ATOMIC_RELEASE,
                                       __HIP_MEMORY_SCOPE_AGENT);
            if (lane < 16) {
                const float* xt = x + (size_t)t * D_;
#pragma unroll
                for (int v = 0; v < 8; ++v) {
                    float xv  = xt[(size_t)v * T_ * D_ + col0 + lane];
                    float sig = 1.0f / (1.0f + __expf(-xv));
                    out[((size_t)v * T_ + t) * D_ + col0 + lane] = hv[v] * (xv * sig);
                    if (t == T_ - 1)
                        out[(size_t)B_ * T_ * D_ + (size_t)v * D_ + col0 + lane] = hv[v];
                }
            }
        }

        // ---- precompute x-projection partial for t+1 (hides its latency) ----
        acc = vzero;
        if (t + 1 < T_) {
            const float* xt1 = x + (size_t)(t + 1) * D_;
#pragma unroll
            for (int kt = 0; kt < D_ / NW; kt += 32) {
                v16bf a  = load_f32frag_g(xt1, (size_t)T_ * D_, ln, k0 + kt, lh);
                v16bf bb = load_wfrag(sWx, ln, k0 + kt, lh);
                acc = __builtin_amdgcn_wmma_f32_16x16x32_bf16(false, a, false, bb,
                                                              (short)0, acc, false, false);
            }
        }
        __syncthreads();   // guard sC reuse across iterations
    }
}

extern "C" void kernel_launch(void* const* d_in, const int* in_sizes, int n_in,
                              void* d_out, int out_size, void* d_ws, size_t ws_size,
                              hipStream_t stream) {
    const float* x  = (const float*)d_in[0];
    const float* h0 = (const float*)d_in[1];
    const float* Wx = (const float*)d_in[2];
    const float* Wh = (const float*)d_in[3];
    const float* b  = (const float*)d_in[4];
    float* out = (float*)d_out;

    unsigned* cnt  = (unsigned*)d_ws;                                        // T_ * 4 B
    __bf16*   hbuf = (__bf16*)((char*)d_ws + (size_t)T_ * sizeof(unsigned)); // 2*B_*D_*2 B

    init_cnt_kernel<<<(T_ + 255) / 256, 256, 0, stream>>>(cnt);
    elman_persistent_kernel<<<G_, NT, 0, stream>>>(x, h0, Wx, Wh, b, out, cnt, hbuf);
}
